// modals_81475529605257
// MI455X (gfx1250) — matrix operand order repaired and checked by reading, not compiled, not run on hardware
//
#include <hip/hip_runtime.h>
#include <hip/hip_bf16.h>

// MI455X (gfx1250) 2-layer LSTM + dense head — persistent-kernel version.
//
// Cost model: recurrence compute per step is tiny (48-64 serial WMMAs/wave),
// so the old 1024-launch design is dispatch-bound. Here the whole recurrence
// is ONE persistent kernel: 64 blocks (32 per layer, layer1 pipelined one
// step behind layer0), software device-wide barrier between phases.
//  - Weights are staged once into LDS (<=64KB per block; one hg-group per
//    block, all 8 waves share them) -> inner loop is ds_load_b128 + v_wmma.
//  - Cell state c lives in registers for all 512 steps.
//  - Only h crosses blocks: double-buffered global fragments, fenced around
//    the barrier.
// Head (z-diff + 2x dense-relu + softmax) stays as tiny VALU kernels.

#define B_   256
#define T_   512
#define D_   128
#define H_   256
#define NBLK 64          // 32 blocks layer0 + 32 blocks layer1

typedef __attribute__((ext_vector_type(16))) _Float16 v16h;
typedef __attribute__((ext_vector_type(8)))  float    v8f;

static __device__ inline v16h zero_v16h() {
  v16h r;
#pragma unroll
  for (int i = 0; i < 16; ++i) r[i] = (_Float16)0;
  return r;
}

static __device__ inline float sigm(float x) {
  return 1.0f / (1.0f + __expf(-x));
}

// ---------------------------------------------------------------------------
// Repack W [K, 4H] (f32, row-major) -> B-fragment f16 layout:
//   Wp[nt][kf][lane][e], lane: n_local = lane&15, kbase = (lane<16)?0:8,
//   elem e: k_local = kbase + e + (e>=8 ? 8 : 0).
__global__ __launch_bounds__(256) void repack_w_kernel(
    const float* __restrict__ W, _Float16* __restrict__ Wp, int K) {
  size_t idx = (size_t)blockIdx.x * blockDim.x + threadIdx.x;
  size_t total = (size_t)K * 1024;
  if (idx >= total) return;
  int e   = (int)(idx & 15);
  int la  = (int)((idx >> 4) & 31);
  int kfr = (int)((idx >> 9) % (size_t)(K >> 5));
  int nt  = (int)(idx / ((size_t)512 * (K >> 5)));
  int kbase = (la < 16) ? 0 : 8;
  int k = kfr * 32 + kbase + e + ((e >= 8) ? 8 : 0);
  int n = nt * 16 + (la & 15);
  Wp[idx] = (_Float16)W[(size_t)k * 1024 + n];
}

// ---------------------------------------------------------------------------
// Repack X [B,T,D] (f32) -> A-fragment f16 layout: Xp[t][mt][kf][lane][e]
//   A 16x32 f16 frag: lane: m = lane&15, khalf = (lane<16)?0:8
//   elem e: v=e/2 ; k_local = (v>=4?16:0) + khalf + 2*(v&3) + (e&1)
__global__ __launch_bounds__(256) void repack_x_kernel(
    const float* __restrict__ X, _Float16* __restrict__ Xp) {
  size_t idx = (size_t)blockIdx.x * blockDim.x + threadIdx.x;  // over B*T*D
  if (idx >= (size_t)B_ * T_ * D_) return;
  int e  = (int)(idx & 15);
  int la = (int)((idx >> 4) & 31);
  int kf = (int)((idx >> 9) & 3);     // D/32 = 4
  int mt = (int)((idx >> 11) & 15);
  int t  = (int)(idx >> 15);
  int b  = mt * 16 + (la & 15);
  int v  = e >> 1;
  int kl = ((v >= 4) ? 16 : 0) + ((la < 16) ? 0 : 8) + 2 * (v & 3) + (e & 1);
  int d  = kf * 32 + kl;
  Xp[idx] = (_Float16)X[((size_t)b * T_ + t) * D_ + d];
}

// ---------------------------------------------------------------------------
__global__ void init_barrier_kernel(unsigned* cnt) {
  if (threadIdx.x == 0 && blockIdx.x == 0) *cnt = 0u;
}

// Device-wide monotonic-counter barrier (all NBLK blocks, co-resident).
static __device__ inline void global_barrier(unsigned* cnt, unsigned target) {
  __syncthreads();
  if (threadIdx.x == 0) {
    __threadfence();  // release: make h/seq writes visible device-wide
    __hip_atomic_fetch_add(cnt, 1u, __ATOMIC_RELEASE, __HIP_MEMORY_SCOPE_AGENT);
    while (__hip_atomic_load(cnt, __ATOMIC_ACQUIRE, __HIP_MEMORY_SCOPE_AGENT) <
           target) {
      __builtin_amdgcn_s_sleep(1);
    }
  }
  __syncthreads();
}

// ---------------------------------------------------------------------------
// Persistent recurrence body for one layer.
// LAYER==0: KXF=4, input = Xp, writes h into seq_out (frag f16, per t).
// LAYER==1: KXF=8, input = hseqp, writes h into z_out (f32 [B,H]).
// Wave mapping: wid = (blockIdx&31)*8 + warp; hg = wid>>4 (shared per block),
// mt = wid&15 -> all 8 waves of a block use the SAME weights (staged in LDS).
template <int LAYER>
static __device__ inline void persist_run(
    const _Float16* __restrict__ xfrag,   // [t][16][KXF][32][16]
    const _Float16* __restrict__ Whp,     // [64][8][32][16]
    const _Float16* __restrict__ Wxp,     // [64][KXF][32][16]
    const float*    __restrict__ bias,    // [1024] order i,f,g,o
    _Float16* __restrict__ hbuf0,         // frag [16][8][32][16]
    _Float16* __restrict__ hbuf1,
    _Float16* __restrict__ seq_out,       // LAYER==0
    float*    __restrict__ z_out,         // LAYER==1
    unsigned* __restrict__ cnt,
    _Float16* lds_w)                      // [4][KT][32][16]
{
  constexpr int KXF = LAYER ? 8 : 4;
  constexpr int KT  = 8 + KXF;
  const int lane = threadIdx.x & 31;
  const int wid  = ((blockIdx.x & 31) << 3) + (threadIdx.x >> 5);
  const int hg = wid >> 4;
  const int mt = wid & 15;

  // ---- stage this hg-group's weights into LDS (once; reused 512 steps) ----
  for (int i = threadIdx.x; i < 4 * KT * 512; i += 256) {
    int e  = i & 15;
    int la = (i >> 4) & 31;
    int kf = (i >> 9) % KT;
    int g  = i / (KT * 512);
    int nt = g * 16 + hg;
    lds_w[i] = (kf < 8)
        ? Whp[(((size_t)nt * 8 + kf) * 32 + la) * 16 + e]
        : Wxp[(((size_t)nt * KXF + (kf - 8)) * 32 + la) * 16 + e];
  }
  __syncthreads();

  float bv[4];
#pragma unroll
  for (int g = 0; g < 4; ++g) bv[g] = bias[g * H_ + hg * 16 + (lane & 15)];

  v8f c;          // cell state: lives in registers for the whole recurrence
#pragma unroll
  for (int r = 0; r < 8; ++r) c[r] = 0.0f;

  _Float16* hb[2] = {hbuf0, hbuf1};

  for (int p = 0; p <= T_; ++p) {
    const bool active = LAYER ? (p >= 1) : (p < T_);
    if (active) {
      const int t = LAYER ? (p - 1) : p;
      const _Float16* h_in  = hb[t & 1];
      _Float16*       h_out = hb[(t + 1) & 1];

      // A fragments: input projection operand for this t
      v16h a_x[KXF];
#pragma unroll
      for (int kf = 0; kf < KXF; ++kf) {
        size_t off = ((((size_t)t * 16 + mt) * KXF + kf) * 32 + lane) * 16;
        a_x[kf] = *(const v16h*)(xfrag + off);
      }
      // A fragments: recurrent operand h_{t-1}
      v16h a_h[8];
      if (t == 0) {
#pragma unroll
        for (int kf = 0; kf < 8; ++kf) a_h[kf] = zero_v16h();
      } else {
#pragma unroll
        for (int kf = 0; kf < 8; ++kf) {
          size_t off = (((size_t)mt * 8 + kf) * 32 + lane) * 16;
          a_h[kf] = *(const v16h*)(h_in + off);
        }
      }

      // ---- 4 gate tiles: bias + h@Wh + x@Wx, weights from LDS ----
      v8f acc[4];
#pragma unroll
      for (int g = 0; g < 4; ++g) {
#pragma unroll
        for (int r = 0; r < 8; ++r) acc[g][r] = bv[g];
      }
#pragma unroll
      for (int g = 0; g < 4; ++g) {
#pragma unroll
        for (int kf = 0; kf < 8; ++kf) {
          v16h bfr = *(const v16h*)(lds_w + (((size_t)g * KT + kf) * 32 + lane) * 16);
          acc[g] = __builtin_amdgcn_wmma_f32_16x16x32_f16(
              false, a_h[kf], false, bfr, (short)0, acc[g], false, false);
        }
#pragma unroll
        for (int kf = 0; kf < KXF; ++kf) {
          v16h bfr = *(const v16h*)(lds_w + (((size_t)g * KT + 8 + kf) * 32 + lane) * 16);
          acc[g] = __builtin_amdgcn_wmma_f32_16x16x32_f16(
              false, a_x[kf], false, bfr, (short)0, acc[g], false, false);
        }
      }

      // ---- fused gate math + c/h update + h scatter to A-frag layout ----
#pragma unroll
      for (int r = 0; r < 8; ++r) {
        float ig = sigm(acc[0][r]);
        float fg = sigm(acc[1][r]);
        float gg = tanhf(acc[2][r]);
        float og = sigm(acc[3][r]);
        float cn = fg * c[r] + ig * gg;
        c[r] = cn;
        float hn = og * tanhf(cn);

        int m_local = r + ((lane < 16) ? 0 : 8);
        int n  = hg * 16 + (lane & 15);       // h column 0..255
        int kf   = n >> 5;
        int kl   = n & 31;
        int hi16 = (kl >= 16) ? 1 : 0;
        int rem  = kl & 15;
        int kh8  = (rem >= 8) ? 1 : 0;
        int rem2 = rem & 7;
        int v = (hi16 ? 4 : 0) + (rem2 >> 1);
        int e = 2 * v + (rem2 & 1);
        int la_dst = m_local + (kh8 ? 16 : 0);
        size_t didx = (((size_t)mt * 8 + kf) * 32 + la_dst) * 16 + e;
        _Float16 hh = (_Float16)hn;
        h_out[didx] = hh;
        if (LAYER == 0) {
          seq_out[(size_t)t * (16 * 8 * 32 * 16) + didx] = hh;
        } else {
          z_out[(size_t)(mt * 16 + m_local) * H_ + n] = hn;
        }
      }
    }
    global_barrier(cnt, (unsigned)(p + 1) * NBLK);
  }
}

__global__ __launch_bounds__(256) void lstm_persistent_kernel(
    const _Float16* __restrict__ Xp, _Float16* __restrict__ hseqp,
    const _Float16* __restrict__ Wx0p, const _Float16* __restrict__ Wh0p,
    const _Float16* __restrict__ Wx1p, const _Float16* __restrict__ Wh1p,
    const float* __restrict__ bl0, const float* __restrict__ bl1,
    _Float16* h0a, _Float16* h0b, _Float16* h1a, _Float16* h1b,
    float* __restrict__ z, unsigned* __restrict__ barrier_cnt) {
  __shared__ _Float16 lds_w[4 * 16 * 32 * 16];  // 64 KB (max: layer1 KT=16)
  if ((blockIdx.x >> 5) == 0) {
    persist_run<0>(Xp, Wh0p, Wx0p, bl0, h0a, h0b, hseqp, nullptr,
                   barrier_cnt, lds_w);
  } else {
    persist_run<1>(hseqp, Wh1p, Wx1p, bl1, h1a, h1b, nullptr, z,
                   barrier_cnt, lds_w);
  }
}

// ---------------------------------------------------------------------------
// Head kernels (tiny; VALU is fine)
__global__ __launch_bounds__(256) void zhat_kernel(
    const float* __restrict__ z, const int* __restrict__ ia,
    const int* __restrict__ ib, float* __restrict__ zh) {
  int tid = blockIdx.x * blockDim.x + threadIdx.x;   // B*H
  if (tid >= B_ * H_) return;
  int b = tid >> 8, h = tid & 255;
  zh[tid] = z[tid] + 0.1f * (z[(size_t)ia[b] * H_ + h] - z[(size_t)ib[b] * H_ + h]);
}

__global__ __launch_bounds__(256) void dense_relu_kernel(
    const float* __restrict__ in, const float* __restrict__ W,
    const float* __restrict__ bias, float* __restrict__ out, int K, int N) {
  int tid = blockIdx.x * blockDim.x + threadIdx.x;   // B*N
  if (tid >= B_ * N) return;
  int b = tid / N, j = tid % N;
  float s = bias[j];
  for (int k = 0; k < K; ++k) s += in[(size_t)b * K + k] * W[(size_t)k * N + j];
  out[tid] = fmaxf(s, 0.0f);
}

__global__ __launch_bounds__(32) void head_softmax_kernel(
    const float* __restrict__ x, const float* __restrict__ W,
    const float* __restrict__ bias, float* __restrict__ out) {
  int b = blockIdx.x;
  int j = threadIdx.x;
  __shared__ float sl[10];
  if (j < 10) {
    float s = bias[j];
    for (int k = 0; k < 256; ++k) s += x[(size_t)b * 256 + k] * W[(size_t)k * 10 + j];
    sl[j] = s;
  }
  __syncthreads();
  if (j < 10) {
    float mx = sl[0];
    for (int q = 1; q < 10; ++q) mx = fmaxf(mx, sl[q]);
    float sum = 0.0f;
    for (int q = 0; q < 10; ++q) sum += __expf(sl[q] - mx);
    out[(size_t)b * 10 + j] = __expf(sl[j] - mx) / sum;
  }
}

// ---------------------------------------------------------------------------
extern "C" void kernel_launch(void* const* d_in, const int* in_sizes, int n_in,
                              void* d_out, int out_size, void* d_ws, size_t ws_size,
                              hipStream_t stream) {
  (void)in_sizes; (void)n_in; (void)out_size; (void)ws_size;
  const float* X   = (const float*)d_in[0];
  const float* Wx0 = (const float*)d_in[1];
  const float* Wh0 = (const float*)d_in[2];
  const float* bl0 = (const float*)d_in[3];
  const float* Wx1 = (const float*)d_in[4];
  const float* Wh1 = (const float*)d_in[5];
  const float* bl1 = (const float*)d_in[6];
  const float* Wd1 = (const float*)d_in[7];
  const float* bd1 = (const float*)d_in[8];
  const float* Wd2 = (const float*)d_in[9];
  const float* bd2 = (const float*)d_in[10];
  const float* Wd3 = (const float*)d_in[11];
  const float* bd3 = (const float*)d_in[12];
  const int*   ia  = (const int*)d_in[13];
  const int*   ib  = (const int*)d_in[14];

  char* ws = (char*)d_ws;
  size_t off = 0;
  auto carve = [&](size_t bytes) {
    size_t r = off;
    off += (bytes + 255) & ~(size_t)255;
    return r;
  };
  _Float16* Xp    = (_Float16*)(ws + carve((size_t)B_ * T_ * D_ * 2));  // 33.5 MB
  _Float16* hseqp = (_Float16*)(ws + carve((size_t)B_ * T_ * H_ * 2));  // 67 MB
  _Float16* Wx0p  = (_Float16*)(ws + carve((size_t)D_ * 4 * H_ * 2));
  _Float16* Wh0p  = (_Float16*)(ws + carve((size_t)H_ * 4 * H_ * 2));
  _Float16* Wx1p  = (_Float16*)(ws + carve((size_t)H_ * 4 * H_ * 2));
  _Float16* Wh1p  = (_Float16*)(ws + carve((size_t)H_ * 4 * H_ * 2));
  _Float16* h0a   = (_Float16*)(ws + carve((size_t)B_ * H_ * 2));
  _Float16* h0b   = (_Float16*)(ws + carve((size_t)B_ * H_ * 2));
  _Float16* h1a   = (_Float16*)(ws + carve((size_t)B_ * H_ * 2));
  _Float16* h1b   = (_Float16*)(ws + carve((size_t)B_ * H_ * 2));
  float*    z     = (float*)(ws + carve((size_t)B_ * H_ * 4));
  float*    zh    = (float*)(ws + carve((size_t)B_ * H_ * 4));
  float*    x1    = (float*)(ws + carve((size_t)B_ * 256 * 4));
  float*    x2    = (float*)(ws + carve((size_t)B_ * 256 * 4));
  unsigned* bcnt  = (unsigned*)(ws + carve(256));

  // One-time repacks into WMMA fragment layouts
  {
    size_t tot = (size_t)B_ * T_ * D_;
    repack_x_kernel<<<(unsigned)((tot + 255) / 256), 256, 0, stream>>>(X, Xp);
  }
  repack_w_kernel<<<(D_ * 1024 + 255) / 256, 256, 0, stream>>>(Wx0, Wx0p, D_);
  repack_w_kernel<<<(H_ * 1024 + 255) / 256, 256, 0, stream>>>(Wh0, Wh0p, H_);
  repack_w_kernel<<<(H_ * 1024 + 255) / 256, 256, 0, stream>>>(Wx1, Wx1p, H_);
  repack_w_kernel<<<(H_ * 1024 + 255) / 256, 256, 0, stream>>>(Wh1, Wh1p, H_);
  init_barrier_kernel<<<1, 32, 0, stream>>>(bcnt);

  // Whole recurrence (both layers, pipelined) in one persistent kernel.
  lstm_persistent_kernel<<<NBLK, 256, 0, stream>>>(
      Xp, hseqp, Wx0p, Wh0p, Wx1p, Wh1p, bl0, bl1,
      h0a, h0b, h1a, h1b, z, bcnt);

  // Head
  zhat_kernel<<<(B_ * H_ + 255) / 256, 256, 0, stream>>>(z, ia, ib, zh);
  dense_relu_kernel<<<(B_ * 256 + 255) / 256, 256, 0, stream>>>(zh, Wd1, bd1, x1, H_, 256);
  dense_relu_kernel<<<(B_ * 256 + 255) / 256, 256, 0, stream>>>(x1, Wd2, bd2, x2, 256, 256);
  head_softmax_kernel<<<B_, 32, 0, stream>>>(x2, Wd3, bd3, (float*)d_out);
}
